// HeteroGraphGCN_33208687133107
// MI455X (gfx1250) — matrix operand order repaired and checked by reading, not compiled, not run on hardware
//
#include <hip/hip_runtime.h>
#include <cstdint>
#include <cstddef>

typedef __attribute__((ext_vector_type(16))) _Float16 v16h;
typedef __attribute__((ext_vector_type(8)))  float    v8f;

#define H 256

__device__ __forceinline__ float gelu_f(float x) {
    return 0.5f * x * (1.0f + erff(x * 0.70710678118654752f));
}

// ---------------- utility kernels ----------------

__global__ void zero_kernel(float* __restrict__ p, size_t n) {
    size_t i = (size_t)blockIdx.x * blockDim.x + threadIdx.x;
    if (i < n) p[i] = 0.0f;
}

// W [K=256][N=256] row-major -> Wt hi/lo [N][K] (transposed, split f32 = hi + lo)
__global__ void splitw_kernel(const float* __restrict__ W,
                              _Float16* __restrict__ hi, _Float16* __restrict__ lo) {
    int i = blockIdx.x * blockDim.x + threadIdx.x;  // 65536
    int j = i >> 8, k = i & 255;
    float v = W[k * H + j];
    _Float16 h = (_Float16)v;
    hi[i] = h;
    lo[i] = (_Float16)(v - (float)h);
}

__global__ void deg_kernel(const int* __restrict__ src, const int* __restrict__ dst,
                           float* __restrict__ dout, float* __restrict__ din, int E) {
    int i = blockIdx.x * blockDim.x + threadIdx.x;
    if (i < E) {
        atomicAdd(&dout[src[i]], 1.0f);
        atomicAdd(&din[dst[i]], 1.0f);
    }
}

__global__ void rsqrtdeg_kernel(float* __restrict__ p, int n) {
    int i = blockIdx.x * blockDim.x + threadIdx.x;
    if (i < n) p[i] = rsqrtf(fmaxf(p[i], 1.0f));
}

// ---------------- WMMA GEMM: out[N,256] = f(A) @ Wt^T ----------------
// A load fused with: optional max-pool(3), optional per-row scale, optional
// per-column affine (BN). Split-f32 via 3x wmma_f32_16x16x32_f16.
__global__ __launch_bounds__(128) void gemm256_kernel(
    const float* __restrict__ A, int ldA, int pool,
    const float* __restrict__ rowscale,          // may be null
    const float* __restrict__ col_a,             // may be null (with col_b)
    const float* __restrict__ col_b,
    const _Float16* __restrict__ Bhi,            // [256 cols][256 k]
    const _Float16* __restrict__ Blo,
    const float* __restrict__ bias,              // may be null
    int do_gelu,
    float* __restrict__ out, int N)
{
    __shared__ __align__(16) _Float16 sAhi[16 * H];
    __shared__ __align__(16) _Float16 sAlo[16 * H];

    const int row0 = blockIdx.x * 16;
    const int tid  = threadIdx.x;

    // Stage A tile (16 rows x 256 k) into LDS as hi/lo f16
    for (int i = tid; i < 16 * H; i += 128) {
        int m = i >> 8, k = i & 255;
        int r = row0 + m;
        float v = 0.0f;
        if (r < N) {
            if (pool) {
                const float* p = A + (size_t)r * ldA + 3 * k;
                v = fmaxf(p[0], fmaxf(p[1], p[2]));
            } else {
                v = A[(size_t)r * ldA + k];
            }
            if (col_a)    v = v * col_a[k] + col_b[k];
            if (rowscale) v *= rowscale[r];
        }
        _Float16 h = (_Float16)v;
        sAhi[i] = h;
        sAlo[i] = (_Float16)(v - (float)h);
    }
    __syncthreads();

    const int wave  = tid >> 5;
    const int lane  = tid & 31;
    const int lhalf = lane >> 4;   // 0 or 1
    const int lmod  = lane & 15;

    v8f acc[4] = {};

    #pragma unroll
    for (int kk = 0; kk < H; kk += 32) {
        // A fragment: M = lmod, K chunks at kk+lhalf*8 and +16 (ISA 16-bit A layout)
        v16h ahi, alo;
        {
            const _Float16* bh = &sAhi[lmod * H + kk + lhalf * 8];
            const _Float16* bl = &sAlo[lmod * H + kk + lhalf * 8];
            #pragma unroll
            for (int i = 0; i < 8; i++) { ahi[i]     = bh[i];      alo[i]     = bl[i]; }
            #pragma unroll
            for (int i = 0; i < 8; i++) { ahi[8 + i] = bh[16 + i]; alo[8 + i] = bl[16 + i]; }
        }
        #pragma unroll
        for (int t = 0; t < 4; t++) {
            int col = (wave * 4 + t) * 16 + lmod;
            // B fragment: N = lmod, K chunks at kk+lhalf*8 and +16 (same as A)
            const _Float16* bh = Bhi + (size_t)col * H + kk + lhalf * 8;
            const _Float16* bl = Blo + (size_t)col * H + kk + lhalf * 8;
            v16h bhi, blo;
            #pragma unroll
            for (int i = 0; i < 8; i++) { bhi[i] = bh[i]; blo[i] = bl[i]; bhi[8 + i] = bh[16 + i]; blo[8 + i] = bl[16 + i]; }
            acc[t] = __builtin_amdgcn_wmma_f32_16x16x32_f16(false, ahi, false, bhi,
                                                            (short)0, acc[t], false, false);
            acc[t] = __builtin_amdgcn_wmma_f32_16x16x32_f16(false, alo, false, bhi,
                                                            (short)0, acc[t], false, false);
            acc[t] = __builtin_amdgcn_wmma_f32_16x16x32_f16(false, ahi, false, blo,
                                                            (short)0, acc[t], false, false);
        }
    }

    // Epilogue: C/D layout -> M = j + 8*lhalf, N = lmod
    #pragma unroll
    for (int t = 0; t < 4; t++) {
        int n = (wave * 4 + t) * 16 + lmod;
        #pragma unroll
        for (int j = 0; j < 8; j++) {
            int r = row0 + j + 8 * lhalf;
            if (r < N) {
                float v = acc[t][j];
                if (bias)    v += bias[n];
                if (do_gelu) v = gelu_f(v);
                out[(size_t)r * H + n] = v;
            }
        }
    }
}

// ---------------- edge scatter: agg[dst] += x[src], one wave per edge ----------------
__global__ __launch_bounds__(256) void scatter_kernel(
    const float* __restrict__ x, const int* __restrict__ src,
    const int* __restrict__ dst, float* __restrict__ agg, int E)
{
    int wid  = (int)(((size_t)blockIdx.x * blockDim.x + threadIdx.x) >> 5);
    int lane = threadIdx.x & 31;
    if (wid >= E) return;
    int s = src[wid], d = dst[wid];
    const float* xp = x   + (size_t)s * H;
    float*       ap = agg + (size_t)d * H;
    #pragma unroll
    for (int i = 0; i < 8; i++) {
        int k = lane + 32 * i;
        atomicAdd(&ap[k], xp[k]);
    }
}

// ---------------- fused  *rsqrt(deg_in)+b -> GELU -> LayerNorm  (one wave/row, in place) ----
__global__ __launch_bounds__(256) void post_ln_kernel(
    float* __restrict__ buf, const float* __restrict__ rin,
    const float* __restrict__ bias, const float* __restrict__ g,
    const float* __restrict__ b, int N)
{
    int wid  = (int)(((size_t)blockIdx.x * blockDim.x + threadIdx.x) >> 5);
    int lane = threadIdx.x & 31;
    if (wid >= N) return;
    float rs = rin[wid];
    float v[8], s = 0.0f, ss = 0.0f;
    #pragma unroll
    for (int i = 0; i < 8; i++) {
        int k = lane + 32 * i;
        float x = buf[(size_t)wid * H + k] * rs + bias[k];
        x = gelu_f(x);
        v[i] = x; s += x; ss += x * x;
    }
    #pragma unroll
    for (int m = 16; m >= 1; m >>= 1) {
        s  += __shfl_xor(s,  m, 32);
        ss += __shfl_xor(ss, m, 32);
    }
    float mean = s * (1.0f / H);
    float var  = ss * (1.0f / H) - mean * mean;
    float inv  = rsqrtf(var + 1e-5f);
    #pragma unroll
    for (int i = 0; i < 8; i++) {
        int k = lane + 32 * i;
        buf[(size_t)wid * H + k] = (v[i] - mean) * inv * g[k] + b[k];
    }
}

// ---------------- BatchNorm column stats + finalize ----------------
__global__ __launch_bounds__(256) void colstats_kernel(
    const float* __restrict__ x, float* __restrict__ cs, float* __restrict__ css, int N)
{
    int col = threadIdx.x;
    float s = 0.0f, ss = 0.0f;
    for (int r = blockIdx.x; r < N; r += gridDim.x) {
        float v = x[(size_t)r * H + col];
        s += v; ss += v * v;
    }
    atomicAdd(&cs[col], s);
    atomicAdd(&css[col], ss);
}

__global__ void bnfin_kernel(const float* __restrict__ cs, const float* __restrict__ css,
                             const float* __restrict__ g, const float* __restrict__ b,
                             float* __restrict__ ca, float* __restrict__ cb, int N)
{
    int c = threadIdx.x;
    float invN = 1.0f / (float)N;
    float mean = cs[c] * invN;
    float var  = css[c] * invN - mean * mean;
    float a = g[c] * rsqrtf(var + 1e-5f);
    ca[c] = a;
    cb[c] = b[c] - mean * a;
}

// ---------------- final [Nc,256] @ [256,1] + b, one wave per row ----------------
__global__ __launch_bounds__(256) void rowdot_kernel(
    const float* __restrict__ x, const float* __restrict__ w,
    const float* __restrict__ bptr, float* __restrict__ out, int N)
{
    int wid  = (int)(((size_t)blockIdx.x * blockDim.x + threadIdx.x) >> 5);
    int lane = threadIdx.x & 31;
    if (wid >= N) return;
    float s = 0.0f;
    #pragma unroll
    for (int i = 0; i < 8; i++) {
        int k = lane + 32 * i;
        s += x[(size_t)wid * H + k] * w[k];
    }
    #pragma unroll
    for (int m = 16; m >= 1; m >>= 1) s += __shfl_xor(s, m, 32);
    if (lane == 0) out[wid] = s + bptr[0];
}

// ---------------- host launch ----------------

extern "C" void kernel_launch(void* const* d_in, const int* in_sizes, int n_in,
                              void* d_out, int out_size, void* d_ws, size_t ws_size,
                              hipStream_t stream) {
    const float* h_claim  = (const float*)d_in[0];
    const int Nc = in_sizes[0] / 768;
    const int Nu = in_sizes[1] / 768;
    const int E0 = in_sizes[4];   // e0_cu edges
    const int E1 = in_sizes[6];   // e1_up edges
    const int* e0_cu_src = (const int*)d_in[4];
    const int* e0_cu_dst = (const int*)d_in[5];
    const int* e1_up_src = (const int*)d_in[6];
    const int* e1_up_dst = (const int*)d_in[7];
    const float* W1_cu  = (const float*)d_in[12];
    const float* b1_cu  = (const float*)d_in[13];
    const float* W2_up  = (const float*)d_in[14];
    const float* b2_up  = (const float*)d_in[15];
    const float* ln_g   = (const float*)d_in[18];
    const float* ln_b   = (const float*)d_in[19];
    const float* bn_g   = (const float*)d_in[20];
    const float* bn_b   = (const float*)d_in[21];
    const float* lin1_W = (const float*)d_in[22];
    const float* lin1_b = (const float*)d_in[23];
    const float* lin2_W = (const float*)d_in[24];
    const float* lin2_b = (const float*)d_in[25];

    uintptr_t base = (uintptr_t)d_ws;
    auto alloc = [&](size_t bytes) -> void* {
        uintptr_t p = (base + 255) & ~(uintptr_t)255;
        base = p + bytes;
        return (void*)p;
    };
    _Float16* w1cu_hi = (_Float16*)alloc(H * H * sizeof(_Float16));
    _Float16* w1cu_lo = (_Float16*)alloc(H * H * sizeof(_Float16));
    _Float16* w2up_hi = (_Float16*)alloc(H * H * sizeof(_Float16));
    _Float16* w2up_lo = (_Float16*)alloc(H * H * sizeof(_Float16));
    _Float16* lin1_hi = (_Float16*)alloc(H * H * sizeof(_Float16));
    _Float16* lin1_lo = (_Float16*)alloc(H * H * sizeof(_Float16));

    const size_t ndeg = 2 * (size_t)Nc + 2 * (size_t)Nu;
    float* degs     = (float*)alloc(ndeg * sizeof(float));
    float* r_cu_out = degs;                 // [Nc]
    float* r_cu_in  = r_cu_out + Nc;        // [Nu]
    float* r_up_out = r_cu_in + Nu;         // [Nu]
    float* r_up_in  = r_up_out + Nu;        // [Nc]

    float* stats  = (float*)alloc(1024 * sizeof(float));
    float* colsum = stats;
    float* colss  = stats + 256;
    float* col_a  = stats + 512;
    float* col_b  = stats + 768;

    const int maxN = (Nc > Nu) ? Nc : Nu;
    float* bufX = (float*)alloc((size_t)maxN * H * sizeof(float)); // x_c, x2, h1
    float* bufU = (float*)alloc((size_t)Nu   * H * sizeof(float)); // agg_u -> u1
    float* bufC = (float*)alloc((size_t)Nc   * H * sizeof(float)); // agg_c -> c2

    auto cdiv = [](long long a, long long b) { return (int)((a + b - 1) / b); };

    // init
    zero_kernel<<<cdiv(ndeg, 256), 256, 0, stream>>>(degs, ndeg);
    zero_kernel<<<2, 256, 0, stream>>>(stats, 512);
    zero_kernel<<<cdiv((size_t)Nu * H, 256), 256, 0, stream>>>(bufU, (size_t)Nu * H);
    zero_kernel<<<cdiv((size_t)Nc * H, 256), 256, 0, stream>>>(bufC, (size_t)Nc * H);

    splitw_kernel<<<256, 256, 0, stream>>>(W1_cu,  w1cu_hi, w1cu_lo);
    splitw_kernel<<<256, 256, 0, stream>>>(W2_up,  w2up_hi, w2up_lo);
    splitw_kernel<<<256, 256, 0, stream>>>(lin1_W, lin1_hi, lin1_lo);

    // degrees -> rsqrt(clip(deg,1))
    deg_kernel<<<cdiv(E0, 256), 256, 0, stream>>>(e0_cu_src, e0_cu_dst, r_cu_out, r_cu_in, E0);
    deg_kernel<<<cdiv(E1, 256), 256, 0, stream>>>(e1_up_src, e1_up_dst, r_up_out, r_up_in, E1);
    rsqrtdeg_kernel<<<cdiv(ndeg, 256), 256, 0, stream>>>(degs, (int)ndeg);

    // conv claim->user: x_c = (amp(h_claim) * r_out) @ W1_cu   (fused pool+scale+GEMM)
    gemm256_kernel<<<cdiv(Nc, 16), 128, 0, stream>>>(
        h_claim, 768, 1, r_cu_out, nullptr, nullptr,
        w1cu_hi, w1cu_lo, nullptr, 0, bufX, Nc);
    scatter_kernel<<<cdiv((long long)E0 * 32, 256), 256, 0, stream>>>(
        bufX, e0_cu_src, e0_cu_dst, bufU, E0);
    post_ln_kernel<<<cdiv((long long)Nu * 32, 256), 256, 0, stream>>>(
        bufU, r_cu_in, b1_cu, ln_g, ln_b, Nu);   // bufU = u1

    // conv user->claim: x2 = (u1 * r_out) @ W2_up
    gemm256_kernel<<<cdiv(Nu, 16), 128, 0, stream>>>(
        bufU, H, 0, r_up_out, nullptr, nullptr,
        w2up_hi, w2up_lo, nullptr, 0, bufX, Nu);
    scatter_kernel<<<cdiv((long long)E1 * 32, 256), 256, 0, stream>>>(
        bufX, e1_up_src, e1_up_dst, bufC, E1);
    post_ln_kernel<<<cdiv((long long)Nc * 32, 256), 256, 0, stream>>>(
        bufC, r_up_in, b2_up, ln_g, ln_b, Nc);   // bufC = c2

    // classifier: BN (batch stats) -> lin1 + GELU -> lin2
    colstats_kernel<<<512, 256, 0, stream>>>(bufC, colsum, colss, Nc);
    bnfin_kernel<<<1, 256, 0, stream>>>(colsum, colss, bn_g, bn_b, col_a, col_b, Nc);
    gemm256_kernel<<<cdiv(Nc, 16), 128, 0, stream>>>(
        bufC, H, 0, nullptr, col_a, col_b,
        lin1_hi, lin1_lo, lin1_b, 1, bufX, Nc);  // bufX = h1
    rowdot_kernel<<<cdiv((long long)Nc * 32, 256), 256, 0, stream>>>(
        bufX, lin2_W, lin2_b, (float*)d_out, Nc);
}
